// Attention1D_42356967473205
// MI455X (gfx1250) — matrix operand order repaired and checked by reading, not compiled, not run on hardware
//
#include <hip/hip_runtime.h>

// ---------------------------------------------------------------------------
// Attention1D for MI455X (gfx1250): bf16 WMMA pipeline with TDM tile feeding
//   LN(f32->bf16, K-innermost xnT) -> qkv GEMM (TDM double-buffered + WMMA)
//   -> sim+softmax fused (WMMA K=8192, LDS softmax) 
//   -> attn@v (TDM attn->LDS + WMMA) -> out proj (TDM + WMMA, f32+bias)
// Intermediate layouts are chosen so every WMMA fragment load is a
// contiguous 16B chunk and every staged tile is a regular 2D TDM tile.
// ---------------------------------------------------------------------------

typedef unsigned short u16;
typedef unsigned int   u32;

typedef __bf16 v8bf  __attribute__((ext_vector_type(8)));
typedef __bf16 v16bf __attribute__((ext_vector_type(16)));
typedef float  v8f   __attribute__((ext_vector_type(8)));
typedef unsigned int u32x4 __attribute__((ext_vector_type(4)));
typedef int          i32x4 __attribute__((ext_vector_type(4)));
typedef int          i32x8 __attribute__((ext_vector_type(8)));

constexpr int Bsz = 8, C = 512, Nseq = 8192;
constexpr int HEADS = 8, DH = 64, HID = 512, O3 = 1536;
constexpr float EPS = 1e-5f;
constexpr float QSCALE = 0.125f; // 64^-0.5

#if __has_builtin(__builtin_amdgcn_tensor_load_to_lds)
#define HAVE_TDM 1
#else
#define HAVE_TDM 0
#endif

__device__ __forceinline__ u16 f2bf(float f) {
  u32 u = __float_as_uint(f);
  u32 r = (u + 0x7FFFu + ((u >> 16) & 1u)) >> 16; // round-nearest-even
  return (u16)r;
}

union AB16 { v16bf v; v8bf h[2]; };
union V8U  { v8bf v; u16 s[8]; };

__device__ __forceinline__ v8f v8f_zero() {
  v8f z = {0.f, 0.f, 0.f, 0.f, 0.f, 0.f, 0.f, 0.f};
  return z;
}

__device__ __forceinline__ v8f wmma_bf16(const AB16& a, const AB16& b, v8f c) {
  return __builtin_amdgcn_wmma_f32_16x16x32_bf16(false, a.v, false, b.v,
                                                 (short)0, c, false, false);
}

// Issue a TDM load of a 2D bf16 tile (rows x cols, row stride = ld elements)
// from global into LDS. Wave-level op (EXEC ignored); caller pairs with
// s_wait_tensorcnt. D# bit packing per CDNA5 ISA 8.3/8.4; tensor dims are set
// equal to the tile so no OOB path triggers.
__device__ __forceinline__ void tdm_issue_tile_2d_bf16(
    const u16* gsrc, u32 lds_off, u32 rows, u32 cols, u32 ld) {
#if HAVE_TDM
  unsigned long long ga = (unsigned long long)(const void*)gsrc;
  u32x4 g0 = { 1u,                                   // count=1, user mode
               lds_off,                              // lds_addr
               (u32)(ga & 0xFFFFFFFFull),            // global_addr[31:0]
               (u32)((ga >> 32) & 0x1FFFFFFull) | (2u << 30) }; // addr[56:32]|type=2
  i32x8 g1 = { (int)(1u << 16),        // data_size=1 (2 bytes), mask=0
               (int)(cols << 16),      // tensor_dim0[15:0]
               (int)(rows << 16),      // tensor_dim0[31:16]=0 | tensor_dim1[15:0]
               (int)(cols << 16),      // tensor_dim1[31:16]=0 | tile_dim0
               (int)rows,              // tile_dim1 | tile_dim2=0
               (int)ld,                // tensor_dim0_stride[31:0]
               0, 0 };
#if defined(__clang_major__) && (__clang_major__ >= 23)
  i32x8 z8 = {0, 0, 0, 0, 0, 0, 0, 0};
  i32x4 z4 = {0, 0, 0, 0};
  __builtin_amdgcn_tensor_load_to_lds(g0, g1, z4, z4, z8, 0);
#else
  i32x4 z4 = {0, 0, 0, 0};
  __builtin_amdgcn_tensor_load_to_lds(g0, g1, z4, z4, 0);
#endif
#else
  (void)gsrc; (void)lds_off; (void)rows; (void)cols; (void)ld;
#endif
}

__device__ __forceinline__ u32 lds_off_of(const void* p) {
  return (u32)(size_t)p; // flat shared address: low 32 bits = LDS offset
}

// ---------------------------------------------------------------------------
// Weight conversion f32 -> bf16
// ---------------------------------------------------------------------------
__global__ __launch_bounds__(256) void k_cvt_weights(
    const float* __restrict__ wqkv, const float* __restrict__ wout,
    u16* __restrict__ wqkv_bf, u16* __restrict__ wout_bf) {
  int i = blockIdx.x * 256 + threadIdx.x;
  if (i < O3 * C)  wqkv_bf[i] = f2bf(wqkv[i]);
  if (i < C * HID) wout_bf[i] = f2bf(wout[i]);
}

// ---------------------------------------------------------------------------
// Channel LayerNorm over C per (b,n) column; output xnT[b][n][c] bf16.
// Per-thread output is 512 contiguous bf16 -> b128 stores. x (134MB) < L2
// (192MB) so the second read pass hits L2.
// ---------------------------------------------------------------------------
__global__ __launch_bounds__(256) void k_layernorm(
    const float* __restrict__ x, const float* __restrict__ gamma,
    u16* __restrict__ xnT) {
  int idx = blockIdx.x * 256 + threadIdx.x; // flat (b,n)
  int b = idx >> 13, n = idx & (Nseq - 1);
  const float* xb = x + ((size_t)b * C) * Nseq + n;
  float s = 0.f, ss = 0.f;
  for (int c = 0; c < C; ++c) {
    float v = xb[(size_t)c * Nseq];
    s += v; ss += v * v;
  }
  float mean = s * (1.f / C);
  float var  = ss * (1.f / C) - mean * mean;
  float rstd = rsqrtf(var + EPS);
  u16* ob = xnT + ((size_t)b * Nseq + n) * C;
  for (int c0 = 0; c0 < C; c0 += 8) {
    V8U u;
#pragma unroll
    for (int e = 0; e < 8; ++e) {
      float v = (xb[(size_t)(c0 + e) * Nseq] - mean) * rstd * gamma[c0 + e];
      u.s[e] = f2bf(v);
    }
    *(v8bf*)(ob + c0) = u.v;
  }
}

// ---------------------------------------------------------------------------
// GEMM: Cout[b] = A(Mdim x 512 bf16, batch-shared, row-major) *
//                 BmT[b](8192 x 512 bf16, n-major / K-innermost)^T
// Block: 256 thr = 8 waves, tile 128(M) x 128(N), K-step 32.
// TDM path: wave 0 DMAs both 128x32 tiles into the alternate LDS buffer each
// step (tensor_load_to_lds), s_wait_tensorcnt 2 overlaps next-tile DMA with
// current-tile WMMAs. Fallback: per-thread b128 staging.
// ---------------------------------------------------------------------------
__global__ __launch_bounds__(256) void k_gemm(
    const u16* __restrict__ A, const u16* __restrict__ BmT,
    void* __restrict__ Cp, const float* __restrict__ bias,
    int Mdim, int out_f32, int scale_rows, float scale) {
  constexpr int K = 512, BK = 32, NSTEPS = K / BK;

  int t = threadIdx.x;
  int lane = t & 31, w = t >> 5;
  int wm = w >> 2, wn = w & 3;            // 2 x 4 wave grid
  int b = blockIdx.z;
  int m0 = blockIdx.y * 128, n0 = blockIdx.x * 128;
  const u16* Bb = BmT + (size_t)b * Nseq * K;
  const u16* Ab = A + (size_t)m0 * K;
  const u16* Bt = Bb + (size_t)n0 * K;

  int la = lane & 15;
  int ah = (lane < 16) ? 0 : 16;  // A k order used here (contiguous, same as B): lanes 0-15 K=0..15, 16-31 K=16..31
  int bh = (lane < 16) ? 0 : 16;  // B k order used here (contiguous, same as A): lanes 0-15 K=0..15, 16-31 K=16..31

  v8f acc[4][2];
#pragma unroll
  for (int it = 0; it < 4; ++it)
#pragma unroll
    for (int nt = 0; nt < 2; ++nt) acc[it][nt] = v8f_zero();

#if HAVE_TDM
  __shared__ __align__(16) u16 As[2][128 * 32];  // [m][k], double-buffered
  __shared__ __align__(16) u16 BT[2][128 * 32];  // [n][k], double-buffered

  if (w == 0) { // prologue: fill buffer 0
    tdm_issue_tile_2d_bf16(Ab, lds_off_of(&As[0][0]), 128, BK, K);
    tdm_issue_tile_2d_bf16(Bt, lds_off_of(&BT[0][0]), 128, BK, K);
  }
  for (int s = 0; s < NSTEPS; ++s) {
    int cur = s & 1;
    if (w == 0) {
      if (s + 1 < NSTEPS) { // DMA next tiles into the other buffer
        int kk2 = (s + 1) * BK;
        tdm_issue_tile_2d_bf16(Ab + kk2, lds_off_of(&As[cur ^ 1][0]), 128, BK, K);
        tdm_issue_tile_2d_bf16(Bt + kk2, lds_off_of(&BT[cur ^ 1][0]), 128, BK, K);
        __builtin_amdgcn_s_wait_tensorcnt(2); // current buffer's pair landed
      } else {
        __builtin_amdgcn_s_wait_tensorcnt(0);
      }
    }
    __syncthreads();

    AB16 aF[4], bF[2];
#pragma unroll
    for (int it = 0; it < 4; ++it) {
      int rl = wm * 64 + it * 16 + la;
      aF[it].h[0] = *(const v8bf*)&As[cur][rl * 32 + ah];
      aF[it].h[1] = *(const v8bf*)&As[cur][rl * 32 + 8 + ah];
    }
#pragma unroll
    for (int nt = 0; nt < 2; ++nt) {
      int cl = wn * 32 + nt * 16 + la;
      bF[nt].h[0] = *(const v8bf*)&BT[cur][cl * 32 + bh];
      bF[nt].h[1] = *(const v8bf*)&BT[cur][cl * 32 + bh + 8];
    }
#pragma unroll
    for (int it = 0; it < 4; ++it)
#pragma unroll
      for (int nt = 0; nt < 2; ++nt)
        acc[it][nt] = wmma_bf16(aF[it], bF[nt], acc[it][nt]);
    __syncthreads(); // compute done before next step's DMA overwrites 'cur'
  }
#else
  __shared__ __align__(16) u16 As[128 * 32];
  __shared__ __align__(16) u16 BT[128 * 32];
  for (int s = 0; s < NSTEPS; ++s) {
    int kk = s * BK;
#pragma unroll
    for (int i = 0; i < 2; ++i) {
      int item = i * 256 + t;
      int row = item >> 2, seg = item & 3;
      *(v8bf*)&As[row * 32 + seg * 8] =
          *(const v8bf*)(Ab + (size_t)row * K + kk + seg * 8);
      *(v8bf*)&BT[row * 32 + seg * 8] =
          *(const v8bf*)(Bt + (size_t)row * K + kk + seg * 8);
    }
    if (kk + BK < K) {
      __builtin_prefetch(Ab + (size_t)(t >> 1) * K + kk + BK, 0, 3);
      __builtin_prefetch(Bt + (size_t)(t >> 1) * K + kk + BK, 0, 3);
    }
    __syncthreads();
    AB16 aF[4], bF[2];
#pragma unroll
    for (int it = 0; it < 4; ++it) {
      int rl = wm * 64 + it * 16 + la;
      aF[it].h[0] = *(const v8bf*)&As[rl * 32 + ah];
      aF[it].h[1] = *(const v8bf*)&As[rl * 32 + 8 + ah];
    }
#pragma unroll
    for (int nt = 0; nt < 2; ++nt) {
      int cl = wn * 32 + nt * 16 + la;
      bF[nt].h[0] = *(const v8bf*)&BT[cl * 32 + bh];
      bF[nt].h[1] = *(const v8bf*)&BT[cl * 32 + bh + 8];
    }
#pragma unroll
    for (int it = 0; it < 4; ++it)
#pragma unroll
      for (int nt = 0; nt < 2; ++nt)
        acc[it][nt] = wmma_bf16(aF[it], bF[nt], acc[it][nt]);
    __syncthreads();
  }
#endif

  // Epilogue: C/D layout -> lanes 0-15 N=la M=r, lanes 16-31 M=r+8
  size_t cbase = (size_t)b * Mdim * Nseq;
#pragma unroll
  for (int it = 0; it < 4; ++it) {
#pragma unroll
    for (int nt = 0; nt < 2; ++nt) {
      int col = n0 + wn * 32 + nt * 16 + la;
#pragma unroll
      for (int r = 0; r < 8; ++r) {
        int row = m0 + wm * 64 + it * 16 + ((lane < 16) ? r : r + 8);
        float v = acc[it][nt][r];
        if (out_f32) {
          ((float*)Cp)[cbase + (size_t)row * Nseq + col] = v + bias[row];
        } else {
          if (row < scale_rows) v *= scale;
          ((u16*)Cp)[cbase + (size_t)row * Nseq + col] = f2bf(v);
        }
      }
    }
  }
}

// ---------------------------------------------------------------------------
// Fused sim + softmax per (b,m):
//   sim = q(64x8192) * k(64x8192)^T accumulated by 8 waves (WMMA, direct
//   global fragments: contraction dim n is contiguous in both operands),
//   staged to 16KB LDS, then threads 0..63 do the row softmax from LDS and
//   write bf16 attn with packed b128 stores.
// ---------------------------------------------------------------------------
__global__ __launch_bounds__(256) void k_sim_softmax(
    const u16* __restrict__ qkv, u16* __restrict__ attn) {
  __shared__ __align__(16) float sim_s[64 * 64]; // 16 KB
  int bm = blockIdx.x;
  int b = bm >> 3, m = bm & 7;
  int t = threadIdx.x, lane = t & 31, w = t >> 5;
  int it = w >> 1, jt0 = (w & 1) * 2;
  int la = lane & 15;
  int ah = (lane < 16) ? 0 : 16;
  int bh = (lane < 16) ? 0 : 16;

  const u16* q = qkv + ((size_t)b * O3 + (size_t)m * DH) * Nseq;
  const u16* k = qkv + ((size_t)b * O3 + HID + (size_t)m * DH) * Nseq;
  const u16* qr = q + (size_t)(it * 16 + la) * Nseq;
  const u16* k0 = k + (size_t)((jt0 + 0) * 16 + la) * Nseq;
  const u16* k1 = k + (size_t)((jt0 + 1) * 16 + la) * Nseq;

  v8f acc0 = v8f_zero(), acc1 = v8f_zero();
#pragma unroll 2
  for (int n = 0; n < Nseq; n += 32) {
    AB16 a, f0, f1;
    a.h[0]  = *(const v8bf*)(qr + n + ah);
    a.h[1]  = *(const v8bf*)(qr + n + 8 + ah);
    f0.h[0] = *(const v8bf*)(k0 + n + bh);
    f0.h[1] = *(const v8bf*)(k0 + n + bh + 8);
    f1.h[0] = *(const v8bf*)(k1 + n + bh);
    f1.h[1] = *(const v8bf*)(k1 + n + bh + 8);
    acc0 = wmma_bf16(a, f0, acc0);
    acc1 = wmma_bf16(a, f1, acc1);
  }
#pragma unroll
  for (int r = 0; r < 8; ++r) {
    int i = it * 16 + ((lane < 16) ? r : r + 8);
    sim_s[i * 64 + (jt0 + 0) * 16 + la] = acc0[r];
    sim_s[i * 64 + (jt0 + 1) * 16 + la] = acc1[r];
  }
  __syncthreads();

  if (t < 64) { // row softmax from LDS
    const float* row = &sim_s[t * 64];
    float mx = -1e30f;
    for (int j = 0; j < 64; ++j) mx = fmaxf(mx, row[j]);
    float ssum = 0.f;
    for (int j = 0; j < 64; ++j) ssum += __expf(row[j] - mx);
    float inv = 1.f / ssum;
    u16* orow = attn + ((size_t)bm * 64 + t) * 64;
    for (int j0 = 0; j0 < 64; j0 += 8) {
      V8U u;
#pragma unroll
      for (int e = 0; e < 8; ++e) u.s[e] = f2bf(__expf(row[j0 + e] - mx) * inv);
      *(v8bf*)(orow + j0) = u.v;
    }
  }
}

// ---------------------------------------------------------------------------
// out_midT[b][n][c] = (attn(64x64) * v(64 x 8192)) stored K-innermost.
// attn tile DMA'd into LDS by the TDM (wave 0, s_wait_tensorcnt, barrier);
// v tile (64x256) transposed into LDS so B fragments are contiguous.
// ---------------------------------------------------------------------------
__global__ __launch_bounds__(256) void k_attnv(const u16* __restrict__ attn,
                                               const u16* __restrict__ qkv,
                                               u16* __restrict__ out_midT) {
  __shared__ __align__(16) u16 attn_s[64 * 64]; // [i][j], 8 KB
  __shared__ __align__(16) u16 vT[256 * 64];    // [n][j], 32 KB
  int bm = blockIdx.y;
  int b = bm >> 3, m = bm & 7;
  int n0 = blockIdx.x * 256;
  int t = threadIdx.x, lane = t & 31, w = t >> 5;
  int la = lane & 15;
  int ah = (lane < 16) ? 0 : 16;
  int bh = (lane < 16) ? 0 : 16;

  const u16* am = attn + (size_t)bm * 64 * 64;
  const u16* v  = qkv + ((size_t)b * O3 + 2 * HID + (size_t)m * DH) * Nseq;

#if HAVE_TDM
  if (w == 0) { // TDM is a wave-level op; one descriptor per workgroup
    tdm_issue_tile_2d_bf16(am, lds_off_of(&attn_s[0]), 64, 64, 64);
    __builtin_amdgcn_s_wait_tensorcnt(0);
  }
#else
#pragma unroll
  for (int i = 0; i < 2; ++i) {
    int item = i * 256 + t; // 512 x b128 covers 64*64
    *(v8bf*)&attn_s[item * 8] = *(const v8bf*)(am + item * 8);
  }
#endif

  // Stage v[64 x 256] -> vT[256 x 64], coalesced global reads along n
#pragma unroll
  for (int i = 0; i < 8; ++i) {
    int item = i * 256 + t;         // 2048 items = 64 rows x 32 segs
    int j = item >> 5, ns = item & 31;
    V8U u;
    u.v = *(const v8bf*)(v + (size_t)j * Nseq + n0 + ns * 8);
#pragma unroll
    for (int e = 0; e < 8; ++e) vT[(ns * 8 + e) * 64 + j] = u.s[e];
  }
  __syncthreads();

  v8f acc[4][2];
#pragma unroll
  for (int it = 0; it < 4; ++it)
#pragma unroll
    for (int nt = 0; nt < 2; ++nt) acc[it][nt] = v8f_zero();

#pragma unroll
  for (int kk = 0; kk < 64; kk += 32) {
    AB16 aF[4], bF[2];
#pragma unroll
    for (int it = 0; it < 4; ++it) {
      const u16* ar = attn_s + (it * 16 + la) * 64 + kk + ah;
      aF[it].h[0] = *(const v8bf*)ar;
      aF[it].h[1] = *(const v8bf*)(ar + 8);
    }
#pragma unroll
    for (int nt = 0; nt < 2; ++nt) {
      int cl = w * 32 + nt * 16 + la;
      bF[nt].h[0] = *(const v8bf*)&vT[cl * 64 + kk + bh];
      bF[nt].h[1] = *(const v8bf*)&vT[cl * 64 + kk + bh + 8];
    }
#pragma unroll
    for (int it = 0; it < 4; ++it)
#pragma unroll
      for (int nt = 0; nt < 2; ++nt)
        acc[it][nt] = wmma_bf16(aF[it], bF[nt], acc[it][nt]);
  }

  // Epilogue: pack 8 consecutive channels per lane -> single b128 store
  u16* ob = out_midT + (size_t)b * Nseq * HID;
#pragma unroll
  for (int it = 0; it < 4; ++it) {
#pragma unroll
    for (int nt = 0; nt < 2; ++nt) {
      int col = n0 + w * 32 + nt * 16 + la;
      int cbase = m * DH + it * 16 + ((lane < 16) ? 0 : 8);
      V8U u;
#pragma unroll
      for (int r = 0; r < 8; ++r) u.s[r] = f2bf(acc[it][nt][r]);
      *(v8bf*)(ob + (size_t)col * HID + cbase) = u.v;
    }
  }
}

// ---------------------------------------------------------------------------
extern "C" void kernel_launch(void* const* d_in, const int* in_sizes, int n_in,
                              void* d_out, int out_size, void* d_ws, size_t ws_size,
                              hipStream_t stream) {
  const float* x      = (const float*)d_in[0];
  const float* gamma  = (const float*)d_in[1];
  const float* w_qkv  = (const float*)d_in[2];
  const float* w_out  = (const float*)d_in[3];
  const float* b_out  = (const float*)d_in[4];
  float* out = (float*)d_out;
  (void)in_sizes; (void)n_in; (void)out_size; (void)ws_size;

  char* ws = (char*)d_ws;
  size_t off = 0;
  auto take = [&](size_t bytes) -> char* {
    char* p = ws + off;
    off = (off + bytes + 255) & ~(size_t)255;
    return p;
  };
  u16*   xnT      = (u16*)take((size_t)Bsz * Nseq * C * 2);       //  64 MB [b][n][c]
  u16*   wqkv_bf  = (u16*)take((size_t)O3 * C * 2);               // 1.5 MB
  u16*   wout_bf  = (u16*)take((size_t)C * HID * 2);              // 0.5 MB
  u16*   qkv      = (u16*)take((size_t)Bsz * O3 * Nseq * 2);      // 192 MB [b][c][n]
  u16*   attn     = (u16*)take((size_t)Bsz * HEADS * 64 * 64 * 2);
  u16*   out_midT = (u16*)take((size_t)Bsz * Nseq * HID * 2);     //  64 MB [b][n][c]

  k_cvt_weights<<<(O3 * C + 255) / 256, 256, 0, stream>>>(w_qkv, w_out, wqkv_bf, wout_bf);
  k_layernorm<<<(Bsz * Nseq) / 256, 256, 0, stream>>>(x, gamma, xnT);
  // qkv projection: M=1536, rows 0..511 (=q) scaled by d^-0.5, bf16 out [c][n]
  k_gemm<<<dim3(Nseq / 128, O3 / 128, Bsz), 256, 0, stream>>>(
      wqkv_bf, xnT, qkv, nullptr, O3, /*out_f32=*/0, /*scale_rows=*/HID, QSCALE);
  k_sim_softmax<<<Bsz * HEADS, 256, 0, stream>>>(qkv, attn);
  k_attnv<<<dim3(Nseq / 256, Bsz * HEADS), 256, 0, stream>>>(attn, qkv, out_midT);
  // output projection: M=512, f32 out + bias
  k_gemm<<<dim3(Nseq / 128, C / 128, Bsz), 256, 0, stream>>>(
      wout_bf, out_midT, out, b_out, C, /*out_f32=*/1, /*scale_rows=*/0, 1.0f);
}
